// SwitchBalancedGate_13615046328977
// MI455X (gfx1250) — matrix operand order repaired and checked by reading, not compiled, not run on hardware
//
#include <hip/hip_runtime.h>
#include <hip/hip_bf16.h>
#include <math.h>

#define TOKENS 32768
#define DMODEL 4096
#define NEXP   64

typedef __attribute__((ext_vector_type(16))) __bf16          v16bf;
typedef __attribute__((ext_vector_type(8)))  float           v8f;
typedef __attribute__((ext_vector_type(16))) unsigned short  v16u;
typedef __attribute__((ext_vector_type(4)))  float           f32x4;
typedef __attribute__((ext_vector_type(8)))  unsigned short  u16x8;

union ABFrag { v16u u; v16bf b; };

__device__ __forceinline__ unsigned short f32_to_bf16_rne(float f) {
  unsigned int u = __float_as_uint(f);
  unsigned int lsb = (u >> 16) & 1u;
  u += 0x7fffu + lsb;                 // round to nearest even
  return (unsigned short)(u >> 16);
}

// ---------------------------------------------------------------------------
// Kernel 0: w1t[k][e] = bf16(w1[e][k])  (B-operand friendly layout), and zero
// the global reduction buffers (required every launch: graph replays).
// ---------------------------------------------------------------------------
__global__ void __launch_bounds__(256)
prep_kernel(const float* __restrict__ w1, unsigned short* __restrict__ w1t,
            float* __restrict__ imp_sum, unsigned int* __restrict__ load_cnt) {
  int i = blockIdx.x * 256 + threadIdx.x;     // 0 .. DMODEL*NEXP-1
  int k = i >> 6;
  int e = i & 63;
  w1t[i] = f32_to_bf16_rne(w1[(size_t)e * DMODEL + k]);
  if (i < NEXP) { imp_sum[i] = 0.f; load_cnt[i] = 0u; }
}

// ---------------------------------------------------------------------------
// Kernel 1: h = tanh(x @ w1^T) via v_wmma_f32_16x16x32_bf16.
// One wave = 16 tokens x 64 experts (4 N-tiles), K-loop over 4096 in steps of 32.
// A layout (16-bit 16x32): lane<16 -> M=lane, K = {0..7, 16..23};
//                          lane>=16 -> M=lane-16, K = {8..15, 24..31}.
// B layout (32x16): N = lane&15, K set per lane half same as A.
// ---------------------------------------------------------------------------
__global__ void __launch_bounds__(256)
gemm1_tanh_kernel(const float* __restrict__ x,
                  const unsigned short* __restrict__ w1t,
                  float* __restrict__ h) {
  const int lane   = threadIdx.x & 31;
  const int wave   = threadIdx.x >> 5;
  const int token0 = (blockIdx.x * 8 + wave) * 16;

  const int m  = lane & 15;
  const int kb = (lane >> 4) * 8;     // 0 or 8

  v8f acc[4];
  acc[0] = v8f{}; acc[1] = v8f{}; acc[2] = v8f{}; acc[3] = v8f{};

  const float* xr = x + (size_t)(token0 + m) * DMODEL + kb;

  for (int k0 = 0; k0 < DMODEL; k0 += 32) {
    __builtin_prefetch(xr + k0 + 128, 0, 1);   // global_prefetch_b8, next chunk

    // ---- A fragment: 16 bf16 per lane, converted from f32 on the fly ----
    ABFrag a;
    f32x4 lo0 = *(const f32x4*)(xr + k0);
    f32x4 lo1 = *(const f32x4*)(xr + k0 + 4);
    f32x4 hi0 = *(const f32x4*)(xr + k0 + 16);
    f32x4 hi1 = *(const f32x4*)(xr + k0 + 20);
#pragma unroll
    for (int i = 0; i < 4; ++i) {
      a.u[i]      = f32_to_bf16_rne(lo0[i]);
      a.u[4 + i]  = f32_to_bf16_rne(lo1[i]);
      a.u[8 + i]  = f32_to_bf16_rne(hi0[i]);
      a.u[12 + i] = f32_to_bf16_rne(hi1[i]);
    }

    // ---- B fragments: N = lane&15, K = {kb..kb+7, 16+kb..16+kb+7} (same as A) ----
    const unsigned short* brow = w1t + (size_t)(k0 + kb) * NEXP + (lane & 15);
#pragma unroll
    for (int j = 0; j < 4; ++j) {
      ABFrag b;
#pragma unroll
      for (int i = 0; i < 8; ++i) { b.u[i] = brow[i * NEXP + 16 * j]; b.u[8 + i] = brow[(16 + i) * NEXP + 16 * j]; }

      acc[j] = __builtin_amdgcn_wmma_f32_16x16x32_bf16(
                   false, a.b, false, b.b, (short)0, acc[j], false, false);
    }
  }

  // C/D layout: VGPR r, lanes 0-15 -> M=r; lanes 16-31 -> M=8+r; N = lane&15
  const int mbase = (lane >> 4) * 8;
  const int n     = lane & 15;
#pragma unroll
  for (int j = 0; j < 4; ++j) {
#pragma unroll
    for (int r = 0; r < 8; ++r) {
      int token = token0 + mbase + r;
      h[(size_t)token * NEXP + 16 * j + n] = tanhf(acc[j][r]);
    }
  }
}

// ---------------------------------------------------------------------------
// Kernel 2: per-token gate: logits = h_row @ w2^T (w2 in LDS), softmax,
// gumbel top-1 (first max wins, like jnp.argmax), block reductions.
// ---------------------------------------------------------------------------
__global__ void __launch_bounds__(128)
gate_kernel(const float* __restrict__ h, const float* __restrict__ w2,
            const float* __restrict__ noise,
            float* __restrict__ out_idx, float* __restrict__ out_score,
            float* __restrict__ imp_sum, unsigned int* __restrict__ load_cnt) {
  __shared__ float        w2s[NEXP * NEXP];   // 16 KB
  __shared__ float        sc[128 * NEXP];     // 32 KB
  __shared__ unsigned int hist[NEXP];

  const int tid = threadIdx.x;
  const int t   = blockIdx.x * 128 + tid;

  for (int i = tid; i < NEXP * NEXP; i += 128) w2s[i] = w2[i];
  if (tid < NEXP) hist[tid] = 0u;
  __syncthreads();

  float hr[NEXP];
  const float* hrow = h + (size_t)t * NEXP;
#pragma unroll
  for (int i = 0; i < NEXP / 4; ++i) {
    f32x4 v = *(const f32x4*)(hrow + 4 * i);
    hr[4*i] = v[0]; hr[4*i+1] = v[1]; hr[4*i+2] = v[2]; hr[4*i+3] = v[3];
  }

  float logits[NEXP];
#pragma unroll 4
  for (int e = 0; e < NEXP; ++e) {
    const float* wr = &w2s[e * NEXP];
    float d = 0.f;
#pragma unroll
    for (int k = 0; k < NEXP; ++k) d = fmaf(hr[k], wr[k], d);
    logits[e] = d;
  }

  // softmax (stable)
  float mx = logits[0];
#pragma unroll
  for (int e = 1; e < NEXP; ++e) mx = fmaxf(mx, logits[e]);
  float s = 0.f;
#pragma unroll
  for (int e = 0; e < NEXP; ++e) {
    float ex = expf(logits[e] - mx);
    s += ex;
    sc[tid * NEXP + e] = ex;
  }
  float inv = 1.f / s;
#pragma unroll
  for (int e = 0; e < NEXP; ++e) sc[tid * NEXP + e] *= inv;

  // gumbel-perturbed top-1
  const float* nrow = noise + (size_t)t * NEXP;
  float best = logits[0] + nrow[0];
  int   bi   = 0;
#pragma unroll
  for (int e = 1; e < NEXP; ++e) {
    float v = logits[e] + nrow[e];
    if (v > best) { best = v; bi = e; }
  }
  out_idx[t]   = (float)bi;
  out_score[t] = best;
  atomicAdd(&hist[bi], 1u);
  __syncthreads();

  // column sums: lane e reads bank (e) each step -> conflict-free
  if (tid < NEXP) {
    float sum = 0.f;
    for (int r = 0; r < 128; ++r) sum += sc[r * NEXP + tid];
    atomicAdd(&imp_sum[tid], sum);
    atomicAdd(&load_cnt[tid], hist[tid]);
  }
}

// ---------------------------------------------------------------------------
// Kernel 3: means + balance loss, write tuple tail.
// Output layout: [0,T) idx(float), [T,2T) scores, [2T] loss,
//                [2T+1, 2T+65) load_mean, [2T+65, 2T+129) importance_mean
// ---------------------------------------------------------------------------
__global__ void finalize_kernel(const float* __restrict__ imp_sum,
                                const unsigned int* __restrict__ load_cnt,
                                float* __restrict__ out) {
  __shared__ float prod[NEXP];
  const int e = threadIdx.x;
  const float invT = 1.f / (float)TOKENS;
  float lm = (float)load_cnt[e] * invT;
  float im = imp_sum[e] * invT;
  out[2 * TOKENS + 1 + e]        = lm;
  out[2 * TOKENS + 1 + NEXP + e] = im;
  prod[e] = im * lm;
  __syncthreads();
  if (e == 0) {
    float s = 0.f;
    for (int i = 0; i < NEXP; ++i) s += prod[i];
    out[2 * TOKENS] = (float)NEXP * s * 0.1f;
  }
}

// ---------------------------------------------------------------------------
extern "C" void kernel_launch(void* const* d_in, const int* in_sizes, int n_in,
                              void* d_out, int out_size, void* d_ws, size_t ws_size,
                              hipStream_t stream) {
  const float* x     = (const float*)d_in[0];
  const float* w1    = (const float*)d_in[1];
  const float* w2    = (const float*)d_in[2];
  const float* noise = (const float*)d_in[3];
  float* out = (float*)d_out;

  char* ws = (char*)d_ws;
  unsigned short* w1t     = (unsigned short*)ws;                       // 512 KB
  float*          h       = (float*)(ws + 524288);                     // 8 MB
  float*          imp_sum = (float*)(ws + 524288 + 8388608);           // 256 B
  unsigned int*   load_cnt= (unsigned int*)(ws + 524288 + 8388608 + 256);

  hipLaunchKernelGGL(prep_kernel, dim3((DMODEL * NEXP) / 256), dim3(256), 0, stream,
                     w1, w1t, imp_sum, load_cnt);
  hipLaunchKernelGGL(gemm1_tanh_kernel, dim3(TOKENS / 128), dim3(256), 0, stream,
                     x, w1t, h);
  hipLaunchKernelGGL(gate_kernel, dim3(TOKENS / 128), dim3(128), 0, stream,
                     h, w2, noise, out, out + TOKENS, imp_sum, load_cnt);
  hipLaunchKernelGGL(finalize_kernel, dim3(1), dim3(NEXP), 0, stream,
                     imp_sum, load_cnt, out);
}
